// SchNetModel_16329465660182
// MI455X (gfx1250) — matrix operand order repaired and checked by reading, not compiled, not run on hardware
//
#include <hip/hip_runtime.h>
#include <hip/hip_bf16.h>

// SchNet forward, MI455X (gfx1250, wave32, WMMA bf16 + TDM/async-LDS staging).
// N=50000 nodes, E=800000 edges, H=128, L=4 layers, G=2000 graphs.

#define N_NODES 50000
#define N_EDGES 800000
#define FEATN   16
#define HDIM    128
#define LAYERS  4
#define GRAPHS  2000

typedef __attribute__((ext_vector_type(16))) __bf16       v16bf;
typedef __attribute__((ext_vector_type(8)))  float        v8f;
typedef __attribute__((ext_vector_type(4)))  unsigned int u32x4;
typedef __attribute__((ext_vector_type(8)))  int          i32x8;
typedef __attribute__((ext_vector_type(4)))  int          i32x4;

union Frag {
    u32x4  u[2];
    v16bf  v;
    __bf16 e[16];
};

// Fast softplus using raw TRANS ops (v_exp_f32 = 2^x, v_log_f32 = log2):
// softplus(x) = ln2 * log2(1 + 2^(x*log2e)); TRANS co-executes with WMMA.
__device__ __forceinline__ float softplus_f(float x) {
#if __has_builtin(__builtin_amdgcn_exp2f) && __has_builtin(__builtin_amdgcn_logf)
    const float e  = __builtin_amdgcn_exp2f(x * 1.4426950408889634f);
    const float sp = 0.6931471805599453f * __builtin_amdgcn_logf(1.f + e);
#else
    const float sp = __logf(1.f + __expf(x));
#endif
    return x > 20.f ? x : sp;
}

__device__ __forceinline__ v8f wmma_bf16(v16bf a, v16bf b, v8f c) {
    // D = A(16x32 bf16) * B(32x16 bf16) + C(16x16 f32)
    return __builtin_amdgcn_wmma_f32_16x16x32_bf16(
        /*neg_a=*/false, a, /*neg_b=*/false, b,
        /*c_mod=*/(short)0, c, /*reuse_a=*/false, /*reuse_b=*/false);
}

// ---------------------------------------------------------------------------
// Asynchronous global->LDS copy (ASYNCcnt path), per-lane 16B.
// LDS address comes from the vdst VGPR; generic LDS pointers keep offset in [31:0].
// ---------------------------------------------------------------------------
__device__ __forceinline__ void async_copy_b128(const void* g, void* l) {
    const unsigned lofs = (unsigned)(unsigned long long)l;
    asm volatile("global_load_async_to_lds_b128 %0, %1, off"
                 :: "v"(lofs), "v"(g) : "memory");
}
__device__ __forceinline__ void async_wait0() {
    asm volatile("s_wait_asynccnt 0" ::: "memory");
}

// ---------------------------------------------------------------------------
// Tensor Data Mover staging of one packed 128x128 bf16 B matrix (32 KB,
// contiguous) into LDS. D# built per ISA 08_async_tensor §8; 2D tile,
// data_size=2B, dims 128x128, stride 128. Issued once per block (TDM ignores
// EXEC); completion via TENSORcnt then workgroup barrier.
// This toolchain's builtin is the 6-arg form:
//   (uint32x4 g0, int32x8 g1, int32x4 g2, int32x4 g3, int32x8, i32 cpol)
// ---------------------------------------------------------------------------
__device__ __forceinline__ void stage_B_tdm(const __bf16* gsrc, void* ldst, int tid) {
#if __has_builtin(__builtin_amdgcn_tensor_load_to_lds) && \
    __has_builtin(__builtin_amdgcn_s_wait_tensorcnt)
    if (tid == 0) {
        const unsigned long long ga = (unsigned long long)gsrc;
        u32x4 g0;
        g0.x = 1u;                                     // count=1, user mode
        g0.y = (unsigned)(unsigned long long)ldst;     // lds_addr
        g0.z = (unsigned)(ga & 0xffffffffu);           // global_addr[31:0]
        g0.w = (unsigned)((ga >> 32) & 0x1ffffffu)     // global_addr[56:32]
               | (2u << 30);                           // type=2 (image)
        i32x8 g1;
        g1[0] = (int)(1u << 16);       // workgroup_mask=0, data_size=1 (2B)
        g1[1] = (int)(128u << 16);     // tensor_dim0 = 128 (bits 79:48 low half)
        g1[2] = (int)(128u << 16);     // tensor_dim1 = 128 (bits 111:80 low half)
        g1[3] = (int)(128u << 16);     // tile_dim0   = 128 (bits 127:112)
        g1[4] = 128;                   // tile_dim1   = 128 (bits 143:128)
        g1[5] = 128;                   // tensor_dim0_stride = 128 (bits 207:160)
        g1[6] = (int)(16384u << 16);   // tensor_dim1_stride = 16384 (bits 255:208)
        g1[7] = 0;
        const i32x4 z4 = {0, 0, 0, 0};
        const i32x8 z8 = {0, 0, 0, 0, 0, 0, 0, 0};
        __builtin_amdgcn_tensor_load_to_lds(g0, g1, z4, z4, z8, 0);
    }
    __builtin_amdgcn_s_wait_tensorcnt(0);
#else
    const u32x4* bg = (const u32x4*)gsrc;
    u32x4* sb = (u32x4*)ldst;
#pragma unroll
    for (int i = 0; i < 8; ++i) sb[tid * 8 + i] = bg[tid * 8 + i];
#endif
    __syncthreads();
}

// ---------------------------------------------------------------------------
// Small prep kernels
// ---------------------------------------------------------------------------

// Wc[l] = de_W2 @ mlp_W1[l]   (128x128 each)
__global__ void compose_wc_kernel(const float* __restrict__ deW2,
                                  const float* __restrict__ mlpW1,
                                  float* __restrict__ Wc) {
    const int i = blockIdx.x, l = blockIdx.y, j = threadIdx.x;
    float acc = 0.f;
    for (int k = 0; k < HDIM; ++k)
        acc += deW2[i * HDIM + k] * mlpW1[(size_t)l * HDIM * HDIM + k * HDIM + j];
    Wc[(size_t)l * HDIM * HDIM + i * HDIM + j] = acc;
}

// bc[l] = de_b2 @ mlp_W1[l] + mlp_b1[l]
__global__ void compose_bc_kernel(const float* __restrict__ deb2,
                                  const float* __restrict__ mlpW1,
                                  const float* __restrict__ mlpb1,
                                  float* __restrict__ bc) {
    const int l = blockIdx.x, j = threadIdx.x;
    float acc = mlpb1[l * HDIM + j];
    for (int k = 0; k < HDIM; ++k)
        acc += deb2[k] * mlpW1[(size_t)l * HDIM * HDIM + k * HDIM + j];
    bc[l * HDIM + j] = acc;
}

// Pack a 128x128 f32 row-major (KxN) matrix into bf16 WMMA-B fragment layout:
// dst[((kc2*8+nt)*32 + lane)*16 + idx], lane = (n&15) | ((k>>3 &1)<<4), idx = (k&7) + ((k>>4 &1)<<3)
__global__ __launch_bounds__(256) void pack_b_kernel(const float* __restrict__ src,
                                                     __bf16* __restrict__ dst) {
    src += (size_t)blockIdx.x * HDIM * HDIM;
    dst += (size_t)blockIdx.x * HDIM * HDIM;
    const int tid = threadIdx.x;
    for (int i = 0; i < 64; ++i) {
        const int linear = i * 256 + tid;
        const int k = linear >> 7, n = linear & 127;
        const int kc2 = k >> 5, kk = k & 31, nt = n >> 4;
        const int lane = (n & 15) | (((kk >> 3) & 1) << 4);
        const int idx  = (kk & 7) + ((kk >> 4) << 3);
        dst[(size_t)(((kc2 * 8 + nt) * 32) + lane) * 16 + idx] = (__bf16)src[linear];
    }
}

__global__ void dist_kernel(const int* __restrict__ erow, const int* __restrict__ ecol,
                            const float* __restrict__ pos, float* __restrict__ dist) {
    const int e = blockIdx.x * blockDim.x + threadIdx.x;
    if (e < N_EDGES) {
        const int r = erow[e], c = ecol[e];
        const float dx = pos[r * 3 + 0] - pos[c * 3 + 0];
        const float dy = pos[r * 3 + 1] - pos[c * 3 + 1];
        const float dz = pos[r * 3 + 2] - pos[c * 3 + 2];
        dist[e] = sqrtf(dx * dx + dy * dy + dz * dz);
    }
}

// h = x @ emb_W + emb_b  (K=16, trivial; VALU)
__global__ void embed_kernel(const float* __restrict__ x, const float* __restrict__ W,
                             const float* __restrict__ b,
                             float* __restrict__ h, __bf16* __restrict__ hb) {
    const int gid = blockIdx.x * blockDim.x + threadIdx.x;
    if (gid < N_NODES * HDIM) {
        const int i = gid >> 7, n = gid & 127;
        float acc = b[n];
#pragma unroll
        for (int f = 0; f < FEATN; ++f) acc += x[i * FEATN + f] * W[f * HDIM + n];
        h[gid]  = acc;
        hb[gid] = (__bf16)acc;
    }
}

__global__ void f32_to_bf16_kernel(const float* __restrict__ in,
                                   __bf16* __restrict__ out, int n) {
    const int gid = blockIdx.x * blockDim.x + threadIdx.x;
    if (gid < n) out[gid] = (__bf16)in[gid];
}

__global__ void pool_kernel(const float* __restrict__ h, const int* __restrict__ batch,
                            float* __restrict__ g) {
    const int gid = blockIdx.x * blockDim.x + threadIdx.x;
    if (gid < N_NODES * HDIM) {
        const int i = gid >> 7, n = gid & 127;
        atomicAdd(g + (size_t)batch[i] * HDIM + n, h[gid]);
    }
}

// out[b] = softplus(g[b]@fc_W1+fc_b1) @ fc_W2 + fc_b2
__global__ __launch_bounds__(64) void head_kernel(const float* __restrict__ g,
                                                  const float* __restrict__ fcW1,
                                                  const float* __restrict__ fcb1,
                                                  const float* __restrict__ fcW2,
                                                  const float* __restrict__ fcb2,
                                                  float* __restrict__ out) {
    __shared__ float red[64];
    const int b = blockIdx.x, j = threadIdx.x;
    float acc = fcb1[j];
    for (int k = 0; k < HDIM; ++k) acc += g[(size_t)b * HDIM + k] * fcW1[k * 64 + j];
    red[j] = softplus_f(acc) * fcW2[j];
    __syncthreads();
    for (int s = 32; s > 0; s >>= 1) {
        if (j < s) red[j] += red[j + s];
        __syncthreads();
    }
    if (j == 0) out[b] = red[0] + fcb2[0];
}

// ---------------------------------------------------------------------------
// Node GEMM: out = epilogue(A[rows,128](bf16) @ Bpk + bias)
// MODE 2: hc = acc + bias           (conv transform)
// MODE 3: h  = softplus(h_old + acc + bias); also write bf16 copy (lin+residual)
// Block: 256 thr = 8 waves; block tile 128 rows x 128 cols; wave tile 16x128.
// B staged with global_load_async_to_lds_b128 (ASYNCcnt).
// ---------------------------------------------------------------------------
template <int MODE>
__global__ __launch_bounds__(256) void node_gemm_kernel(
    const __bf16* __restrict__ A, const __bf16* __restrict__ Bpk,
    const float* __restrict__ bias, const float* __restrict__ h_old,
    float* __restrict__ outF, __bf16* __restrict__ outB, int rows) {
    __shared__ u32x4 shB[2048];  // 32 KB packed B
    const int tid = threadIdx.x;
    {
        const u32x4* bg = (const u32x4*)Bpk;
#pragma unroll
        for (int i = 0; i < 8; ++i) async_copy_b128(bg + tid * 8 + i, &shB[tid * 8 + i]);
        async_wait0();
    }
    __syncthreads();

    const int wave = tid >> 5, lane = tid & 31, l15 = lane & 15;
    const int rowBase = blockIdx.x * 128 + wave * 16;
    int arow = rowBase + l15;
    if (arow >= rows) arow = rows - 1;  // clamp loads; stores guarded below
    const __bf16* arp = A + (size_t)arow * HDIM;
    __builtin_prefetch(arp, 0, 1);  // global_prefetch_b8

    v8f acc[8] = {};
    const int kbase8 = (lane >> 4) << 3;  // 0 / 8 : A-layout lane-half K base
#pragma unroll
    for (int kc2 = 0; kc2 < 4; ++kc2) {
        Frag a;
        const int k0 = kc2 * 32 + kbase8;
        a.u[0] = *(const u32x4*)(arp + k0);
        a.u[1] = *(const u32x4*)(arp + k0 + 16);
#pragma unroll
        for (int nt = 0; nt < 8; ++nt) {
            Frag b;
            const int fo = ((kc2 * 8 + nt) * 32 + lane) * 2;
            b.u[0] = shB[fo];
            b.u[1] = shB[fo + 1];
            acc[nt] = wmma_bf16(a.v, b.v, acc[nt]);
        }
    }

    const int mHalf = (lane >> 4) << 3;  // C layout: M = v + 8*(lane>=16)
#pragma unroll
    for (int nt = 0; nt < 8; ++nt) {
        const int n = nt * 16 + l15;
        const float bn = bias[n];
#pragma unroll
        for (int v = 0; v < 8; ++v) {
            const int r = rowBase + v + mHalf;
            if (r < rows) {
                const size_t idx = (size_t)r * HDIM + n;
                const float val = acc[nt][v] + bn;
                if (MODE == 2) {
                    outF[idx] = val;
                } else {
                    const float hv = softplus_f(h_old[idx] + val);
                    outF[idx] = hv;
                    outB[idx] = (__bf16)hv;
                }
            }
        }
    }
}

// ---------------------------------------------------------------------------
// Fused edge kernel (per layer):
//   s    = softplus(dist * de_W1 + de_b1)         built per-fragment in regs
//   t    = softplus(s @ Wc + bc)                  GEMM1 -> LDS (A-frag layout)
//   m    = t @ W2 + b2                            GEMM2 from LDS
//   agg[col] += m * hc[row]                       gather + atomic scatter
// Block tile: 128 edges x 128 cols. LDS: 32KB B-stage (TDM, reused) + 32KB t.
// ---------------------------------------------------------------------------
__global__ __launch_bounds__(256) void edge_layer_kernel(
    const float* __restrict__ dist, const int* __restrict__ erow,
    const int* __restrict__ ecol, const float* __restrict__ w1,
    const float* __restrict__ b1, const __bf16* __restrict__ B1pk,
    const float* __restrict__ bc, const __bf16* __restrict__ B2pk,
    const float* __restrict__ b2, const float* __restrict__ hc,
    float* __restrict__ agg) {
    __shared__ u32x4 shB[2048];  // 32 KB: Wc packed, then W2 packed
    __shared__ u32x4 shT[2048];  // 32 KB: t in A-fragment layout (4KB per wave)
    const int tid = threadIdx.x;
    const int wave = tid >> 5, lane = tid & 31, l15 = lane & 15;
    const int eBase = blockIdx.x * 128 + wave * 16;
    const float d = dist[eBase + l15];  // E is a multiple of 128

    stage_B_tdm(B1pk, shB, tid);  // TENSOR_LOAD_TO_LDS + s_wait_tensorcnt + barrier

    // ---- GEMM1: t = softplus(s @ Wc + bc), s synthesized from dist ----
    v8f acc[8] = {};
    const int kbase8 = (lane >> 4) << 3;
#pragma unroll
    for (int kc2 = 0; kc2 < 4; ++kc2) {
        Frag a;
#pragma unroll
        for (int j = 0; j < 16; ++j) {
            // A 16x32 bf16 layout: lanes0-15 K={0..7,16..23}, lanes16-31 K={8..15,24..31}
            const int k = kc2 * 32 + (j + (j & 8) + kbase8);
            a.e[j] = (__bf16)softplus_f(d * w1[k] + b1[k]);
        }
#pragma unroll
        for (int nt = 0; nt < 8; ++nt) {
            Frag b;
            const int fo = ((kc2 * 8 + nt) * 32 + lane) * 2;
            b.u[0] = shB[fo];
            b.u[1] = shB[fo + 1];
            acc[nt] = wmma_bf16(a.v, b.v, acc[nt]);
        }
    }

    // ---- store t into shT in packed A-fragment layout (wave-local 4KB) ----
    __bf16* tp = (__bf16*)shT;
    const int mHalf = (lane >> 4) << 3;
#pragma unroll
    for (int nt = 0; nt < 8; ++nt) {
        const int n = nt * 16 + l15;  // K index of GEMM2
        const int kk = n & 31;
        const int kc2n = n >> 5;
        const int lgrp = (kk >> 3) & 1;
        const int jidx = (kk & 7) + ((kk >= 16) ? 8 : 0);
        const float bn = bc[n];
#pragma unroll
        for (int v = 0; v < 8; ++v) {
            const int m = v + mHalf;  // edge row within wave tile
            const float tv = softplus_f(acc[nt][v] + bn);
            tp[wave * 2048 + kc2n * 512 + (m + 16 * lgrp) * 16 + jidx] = (__bf16)tv;
        }
    }
    __syncthreads();  // all GEMM1 B reads + t writes done

    stage_B_tdm(B2pk, shB, tid);  // swap W2 into the same staging region

    // ---- GEMM2: m = t @ W2 + b2 ----
    v8f acc2[8] = {};
#pragma unroll
    for (int kc2 = 0; kc2 < 4; ++kc2) {
        Frag a;
        const int ao = wave * 256 + kc2 * 64 + lane * 2;  // u32x4 units
        a.u[0] = shT[ao];
        a.u[1] = shT[ao + 1];
#pragma unroll
        for (int nt = 0; nt < 8; ++nt) {
            Frag b;
            const int fo = ((kc2 * 8 + nt) * 32 + lane) * 2;
            b.u[0] = shB[fo];
            b.u[1] = shB[fo + 1];
            acc2[nt] = wmma_bf16(a.v, b.v, acc2[nt]);
        }
    }

    // ---- epilogue: gather hc[row], scatter-add into agg[col] ----
    int rr[8], cc[8];
#pragma unroll
    for (int v = 0; v < 8; ++v) {
        const int ee = eBase + v + mHalf;
        rr[v] = erow[ee];
        cc[v] = ecol[ee];
    }
#pragma unroll
    for (int nt = 0; nt < 8; ++nt) {
        const int n = nt * 16 + l15;
        const float bn = b2[n];
#pragma unroll
        for (int v = 0; v < 8; ++v) {
            const float m = acc2[nt][v] + bn;
            const float val = m * hc[(size_t)rr[v] * HDIM + n];
            atomicAdd(agg + (size_t)cc[v] * HDIM + n, val);
        }
    }
}

// ---------------------------------------------------------------------------
// Host orchestration
// ---------------------------------------------------------------------------
extern "C" void kernel_launch(void* const* d_in, const int* in_sizes, int n_in,
                              void* d_out, int out_size, void* d_ws, size_t ws_size,
                              hipStream_t stream) {
    const float* x      = (const float*)d_in[0];
    const int*   ei     = (const int*)d_in[1];   // [2,E]
    const int*   batch  = (const int*)d_in[2];
    const float* pos    = (const float*)d_in[3];
    const float* emb_W  = (const float*)d_in[4];
    const float* emb_b  = (const float*)d_in[5];
    const float* de_W1  = (const float*)d_in[6];
    const float* de_b1  = (const float*)d_in[7];
    const float* de_W2  = (const float*)d_in[8];
    const float* de_b2  = (const float*)d_in[9];
    const float* mlp_W1 = (const float*)d_in[10];
    const float* mlp_b1 = (const float*)d_in[11];
    const float* mlp_W2 = (const float*)d_in[12];
    const float* mlp_b2 = (const float*)d_in[13];
    const float* conv_W = (const float*)d_in[14];
    const float* conv_b = (const float*)d_in[15];
    const float* lin_W  = (const float*)d_in[16];
    const float* lin_b  = (const float*)d_in[17];
    const float* fc_W1  = (const float*)d_in[18];
    const float* fc_b1  = (const float*)d_in[19];
    const float* fc_W2  = (const float*)d_in[20];
    const float* fc_b2  = (const float*)d_in[21];
    float* out = (float*)d_out;

    const int* erow = ei;
    const int* ecol = ei + N_EDGES;

    // workspace carve-out (~106 MB total)
    char* ws = (char*)d_ws;
    size_t off = 0;
    auto carve = [&](size_t bytes) -> char* {
        char* p = ws + off;
        off = (off + bytes + 255) & ~(size_t)255;
        return p;
    };
    float*  h_f32  = (float*)carve((size_t)N_NODES * HDIM * 4);
    __bf16* h_bf   = (__bf16*)carve((size_t)N_NODES * HDIM * 2);
    float*  hc     = (float*)carve((size_t)N_NODES * HDIM * 4);
    float*  agg    = (float*)carve((size_t)N_NODES * HDIM * 4);
    __bf16* agg_bf = (__bf16*)carve((size_t)N_NODES * HDIM * 2);
    float*  dist   = (float*)carve((size_t)N_EDGES * 4);
    float*  gpool  = (float*)carve((size_t)GRAPHS * HDIM * 4);
    float*  Wc     = (float*)carve((size_t)LAYERS * HDIM * HDIM * 4);
    float*  bc     = (float*)carve((size_t)LAYERS * HDIM * 4);
    __bf16* pkWc   = (__bf16*)carve((size_t)LAYERS * HDIM * HDIM * 2);
    __bf16* pkW2   = (__bf16*)carve((size_t)LAYERS * HDIM * HDIM * 2);
    __bf16* pkCW   = (__bf16*)carve((size_t)LAYERS * HDIM * HDIM * 2);
    __bf16* pkLW   = (__bf16*)carve((size_t)LAYERS * HDIM * HDIM * 2);

    // weight prep
    compose_wc_kernel<<<dim3(HDIM, LAYERS), HDIM, 0, stream>>>(de_W2, mlp_W1, Wc);
    compose_bc_kernel<<<LAYERS, HDIM, 0, stream>>>(de_b2, mlp_W1, mlp_b1, bc);
    pack_b_kernel<<<LAYERS, 256, 0, stream>>>(Wc, pkWc);
    pack_b_kernel<<<LAYERS, 256, 0, stream>>>(mlp_W2, pkW2);
    pack_b_kernel<<<LAYERS, 256, 0, stream>>>(conv_W, pkCW);
    pack_b_kernel<<<LAYERS, 256, 0, stream>>>(lin_W, pkLW);

    // geometry + embedding
    dist_kernel<<<(N_EDGES + 255) / 256, 256, 0, stream>>>(erow, ecol, pos, dist);
    embed_kernel<<<(N_NODES * HDIM + 255) / 256, 256, 0, stream>>>(
        x, emb_W, emb_b, h_f32, h_bf);

    const int nodeBlocks = (N_NODES + 127) / 128;  // 391
    const int edgeBlocks = N_EDGES / 128;          // 6250
    const int nhElems = N_NODES * HDIM;

    for (int l = 0; l < LAYERS; ++l) {
        const size_t wOff = (size_t)l * HDIM * HDIM;
        // hc = h @ conv_W[l] + conv_b[l]
        node_gemm_kernel<2><<<nodeBlocks, 256, 0, stream>>>(
            h_bf, pkCW + wOff, conv_b + l * HDIM, nullptr, hc, nullptr, N_NODES);
        (void)hipMemsetAsync(agg, 0, (size_t)nhElems * 4, stream);
        // fused edge MLP + conv-gather + scatter
        edge_layer_kernel<<<edgeBlocks, 256, 0, stream>>>(
            dist, erow, ecol, de_W1, de_b1, pkWc + wOff, bc + l * HDIM,
            pkW2 + wOff, mlp_b2 + l * HDIM, hc, agg);
        f32_to_bf16_kernel<<<(nhElems + 255) / 256, 256, 0, stream>>>(agg, agg_bf, nhElems);
        // h = softplus(h + agg @ lin_W[l] + lin_b[l])
        node_gemm_kernel<3><<<nodeBlocks, 256, 0, stream>>>(
            agg_bf, pkLW + wOff, lin_b + l * HDIM, h_f32, h_f32, h_bf, N_NODES);
    }

    // pooling + head
    (void)hipMemsetAsync(gpool, 0, (size_t)GRAPHS * HDIM * 4, stream);
    pool_kernel<<<(nhElems + 255) / 256, 256, 0, stream>>>(h_f32, batch, gpool);
    head_kernel<<<GRAPHS, 64, 0, stream>>>(gpool, fc_W1, fc_b1, fc_W2, fc_b2, out);
}
